// NeuralNet_498216206732
// MI455X (gfx1250) — matrix operand order repaired and checked by reading, not compiled, not run on hardware
//
#include <hip/hip_runtime.h>
#include <hip/hip_bf16.h>

#define IN_SIZE 3072
#define HIDDEN  32
#define OUT_SIZE 10
#define WAVES_PER_BLOCK 8
#define ROWS_PER_WAVE   16
#define ROWS_PER_BLOCK  (WAVES_PER_BLOCK * ROWS_PER_WAVE)   // 128
#define HSTRIDE 33   // padded LDS row stride (floats) to avoid bank conflicts

typedef __attribute__((ext_vector_type(16))) __bf16 v16bf;
typedef __attribute__((ext_vector_type(8)))  float  v8f;
typedef __attribute__((ext_vector_type(4)))  float  v4f;

__device__ __forceinline__ v4f ld4_nt(const float* p) {
    return __builtin_nontemporal_load((const v4f*)p);
}
__device__ __forceinline__ v4f ld4(const float* p) {
    return *(const v4f*)p;
}

__global__ __launch_bounds__(256) void mlp_wmma_kernel(
    const float* __restrict__ x,
    const float* __restrict__ W1,
    const float* __restrict__ b1,
    const float* __restrict__ W2,
    const float* __restrict__ b2,
    float* __restrict__ out)
{
    __shared__ float w2s[OUT_SIZE * HIDDEN];
    __shared__ float b1s[HIDDEN];
    __shared__ float b2s[OUT_SIZE];
    __shared__ float hbuf[WAVES_PER_BLOCK][ROWS_PER_WAVE * HSTRIDE];

    const int tid = threadIdx.x;
    if (tid < OUT_SIZE * HIDDEN) w2s[tid] = W2[tid];
    if (tid < HIDDEN)            b1s[tid] = b1[tid];
    if (tid < OUT_SIZE)          b2s[tid] = b2[tid];

    const int wave = tid >> 5;
    const int lane = tid & 31;
    const int half = lane >> 4;     // 0: lanes 0-15, 1: lanes 16-31
    const int n16  = lane & 15;

    const int mbase = (blockIdx.x * WAVES_PER_BLOCK + wave) * ROWS_PER_WAVE;

    // A fragment: row = mbase+n16; 16 consecutive K per lane (matches B)
    const int koffA = half ? 16 : 0;
    const float* xrow = x + (size_t)(mbase + n16) * IN_SIZE + koffA;

    // B fragments: column (hidden unit) = n16 (+16 for second tile); 16 consecutive K per lane
    const int koffB = half ? 16 : 0;
    const float* w1a = W1 + (size_t)(n16)      * IN_SIZE + koffB;   // hidden 0-15
    const float* w1b = W1 + (size_t)(n16 + 16) * IN_SIZE + koffB;   // hidden 16-31

    v8f acc0 = {};  // H[mbase.., 0..15]
    v8f acc1 = {};  // H[mbase.., 16..31]

    #pragma unroll 2
    for (int kb = 0; kb < IN_SIZE; kb += 32) {
        // ---- A fragment: x[row, kb+koffA + 0..15] (bf16)
        float ax[16];
        *(v4f*)&ax[0]  = ld4_nt(xrow + kb);
        *(v4f*)&ax[4]  = ld4_nt(xrow + kb + 4);
        *(v4f*)&ax[8]  = ld4_nt(xrow + kb + 8);
        *(v4f*)&ax[12] = ld4_nt(xrow + kb + 12);
        v16bf afrag;
        #pragma unroll
        for (int j = 0; j < 16; ++j) afrag[j] = (__bf16)ax[j];

        // ---- B fragment 0: W1[n16, kb+koffB + 0..15]
        float bx[16];
        *(v4f*)&bx[0]  = ld4(w1a + kb);
        *(v4f*)&bx[4]  = ld4(w1a + kb + 4);
        *(v4f*)&bx[8]  = ld4(w1a + kb + 8);
        *(v4f*)&bx[12] = ld4(w1a + kb + 12);
        v16bf bfrag0;
        #pragma unroll
        for (int j = 0; j < 16; ++j) bfrag0[j] = (__bf16)bx[j];

        // ---- B fragment 1: W1[n16+16, kb+koffB + 0..15]
        float cx[16];
        *(v4f*)&cx[0]  = ld4(w1b + kb);
        *(v4f*)&cx[4]  = ld4(w1b + kb + 4);
        *(v4f*)&cx[8]  = ld4(w1b + kb + 8);
        *(v4f*)&cx[12] = ld4(w1b + kb + 12);
        v16bf bfrag1;
        #pragma unroll
        for (int j = 0; j < 16; ++j) bfrag1[j] = (__bf16)cx[j];

        acc0 = __builtin_amdgcn_wmma_f32_16x16x32_bf16(
                   false, afrag, false, bfrag0, (short)0, acc0, false, false);
        acc1 = __builtin_amdgcn_wmma_f32_16x16x32_bf16(
                   false, afrag, false, bfrag1, (short)0, acc1, false, false);
    }

    __syncthreads();   // w2s/b1s/b2s visible; (also orders the preload stores)

    // ---- bias + ReLU, scatter H tile into padded LDS
    // C/D layout: element v -> row (half*8 + v), col n16 (acc0) / n16+16 (acc1)
    const float bias0 = b1s[n16];
    const float bias1 = b1s[16 + n16];
    float* hrow = &hbuf[wave][0];
    #pragma unroll
    for (int v = 0; v < 8; ++v) {
        const int m = half * 8 + v;
        float h0 = acc0[v] + bias0; h0 = h0 > 0.f ? h0 : 0.f;
        float h1 = acc1[v] + bias1; h1 = h1 > 0.f ? h1 : 0.f;
        hrow[m * HSTRIDE + n16]      = h0;
        hrow[m * HSTRIDE + 16 + n16] = h1;
    }
    __syncthreads();   // make each wave's H tile visible to itself (and block)

    // ---- layer 2: lane handles row r = n16, outputs [obase, obase+5)
    const int r = n16;
    const int obase = half * 5;   // 0-4 or 5-9
    float s0 = 0.f, s1 = 0.f, s2 = 0.f, s3 = 0.f, s4 = 0.f;
    #pragma unroll
    for (int h = 0; h < HIDDEN; ++h) {
        const float hv = hrow[r * HSTRIDE + h];
        s0 += hv * w2s[(obase + 0) * HIDDEN + h];
        s1 += hv * w2s[(obase + 1) * HIDDEN + h];
        s2 += hv * w2s[(obase + 2) * HIDDEN + h];
        s3 += hv * w2s[(obase + 3) * HIDDEN + h];
        s4 += hv * w2s[(obase + 4) * HIDDEN + h];
    }

    float zs[5] = { s0, s1, s2, s3, s4 };
    float* orow = out + (size_t)(mbase + r) * OUT_SIZE + obase;
    #pragma unroll
    for (int j = 0; j < 5; ++j) {
        const float z = zs[j] + b2s[obase + j];
        orow[j] = 1.f / (1.f + __expf(-z));
    }
}

extern "C" void kernel_launch(void* const* d_in, const int* in_sizes, int n_in,
                              void* d_out, int out_size, void* d_ws, size_t ws_size,
                              hipStream_t stream) {
    const float* x  = (const float*)d_in[0];
    const float* W1 = (const float*)d_in[1];
    const float* b1 = (const float*)d_in[2];
    const float* W2 = (const float*)d_in[3];
    const float* b2 = (const float*)d_in[4];
    float* out = (float*)d_out;

    const int n = in_sizes[0] / IN_SIZE;            // 32768
    const int grid = n / ROWS_PER_BLOCK;            // 256 blocks of 8 waves
    mlp_wmma_kernel<<<grid, 256, 0, stream>>>(x, W1, b1, W2, b2, out);
}
